// Block_63402307224451
// MI455X (gfx1250) — matrix operand order repaired and checked by reading, not compiled, not run on hardware
//
#include <hip/hip_runtime.h>
#include <math.h>

// ---------------------------------------------------------------------------
// Transformer block for gfx1250 (MI455X).
//  - GEMMs: v_wmma_f32_16x16x32_f16, LDS double-buffered; A tiles via the
//    Tensor Data Mover (tensor_load_to_lds, TENSORcnt), B tiles via
//    global_load_async_to_lds_b128 (ASYNCcnt).
//  - Attention: flash w/ online softmax; S^T = K.Q^T keeps softmax rows
//    in-lane and P already in WMMA A-layout; V^T staged in LDS.
// ---------------------------------------------------------------------------

typedef _Float16 half_t;
typedef __attribute__((ext_vector_type(16))) _Float16 v16h;
typedef __attribute__((ext_vector_type(8)))  _Float16 v8h;
typedef __attribute__((ext_vector_type(8)))  float    v8f;
typedef int v4i __attribute__((ext_vector_type(4)));
typedef int v8i __attribute__((ext_vector_type(8)));

#define DIM    768
#define HEADS  12
#define HD     64
#define HIDDEN 3072
#define SEQ    1024
#define BATCH  4
#define ROWS   (BATCH * SEQ)   // 4096
#define QKVC   (3 * DIM)       // 2304

// -------- CDNA5 async global->LDS copy (ASYNCcnt-tracked) -------------------
__device__ __forceinline__ void async_b128(unsigned lds_off, const void* g) {
  asm volatile("global_load_async_to_lds_b128 %0, %1, off"
               :: "v"(lds_off), "v"((unsigned long long)(size_t)g)
               : "memory");
}
__device__ __forceinline__ void wait_async0() {
  asm volatile("s_wait_asynccnt 0x0" ::: "memory");
}

// -------- CDNA5 Tensor Data Mover: 2D tile (f16) -> LDS ---------------------
// D# per CDNA5 ISA: group0 = {flags, lds_addr,
// global_addr, type=2}; group1 = {data_size=2B, tensor dims, tile dims,
// dim0 stride}; groups 2/3 NULL (2D tensor).
__device__ __forceinline__ void tdm_load_2d(unsigned lds_off, const void* gaddr,
                                            unsigned tensor_d0, unsigned tensor_d1,
                                            unsigned tile_d0, unsigned tile_d1,
                                            unsigned stride0) {
  unsigned long long ga = (unsigned long long)(size_t)gaddr;
  v4i g0;
  g0[0] = 1;                                           // count=1 (user D#)
  g0[1] = (int)lds_off;                                // lds_addr (bytes)
  g0[2] = (int)(unsigned)(ga & 0xFFFFFFFFu);           // global_addr[31:0]
  g0[3] = (int)((unsigned)(ga >> 32) | 0x80000000u);   // addr[56:32] | type=2
  v8i g1;
  g1[0] = (int)(1u << 16);                             // data_size=1 (2 bytes)
  g1[1] = (int)((tensor_d0 & 0xFFFFu) << 16);          // tensor_dim0[15:0]
  g1[2] = (int)((tensor_d0 >> 16) | ((tensor_d1 & 0xFFFFu) << 16));
  g1[3] = (int)((tensor_d1 >> 16) | (tile_d0 << 16));  // | tile_dim0
  g1[4] = (int)tile_d1;                                // tile_dim1 (tile_dim2=0)
  g1[5] = (int)stride0;                                // tensor_dim0_stride lo
  g1[6] = 0;                                           // stride hi, d1stride lo
  g1[7] = 0;
  asm volatile("tensor_load_to_lds %0, %1" :: "s"(g0), "s"(g1) : "memory");
}

// ---------------- one-time weight cast + transpose (f32 KxN -> f16 NxK) -----
__global__ void cast_transpose_kernel(const float* __restrict__ w,
                                      half_t* __restrict__ wT, int K, int N) {
  size_t idx = (size_t)blockIdx.x * 256 + threadIdx.x;
  if (idx >= (size_t)K * N) return;
  int n = (int)(idx / K);
  int k = (int)(idx % K);
  wT[idx] = (half_t)w[(size_t)k * N + n];
}

// ---------------- mask -> additive bias, computed once ----------------------
__global__ void mask_bias_kernel(const int* __restrict__ mask,
                                 float* __restrict__ bias) {
  int i = blockIdx.x * 256 + threadIdx.x;
  if (i < BATCH * SEQ) bias[i] = (mask[i] == 0) ? -10000.0f : 0.0f;
}

// ---------------- LayerNorm (768 cols) -> f16 -------------------------------
__global__ __launch_bounds__(256)
void layernorm_f16_kernel(const float* __restrict__ x, const float* __restrict__ g,
                          const float* __restrict__ b, half_t* __restrict__ out) {
  const int row = blockIdx.x;
  const int t = threadIdx.x;
  const float* xr = x + (size_t)row * DIM;
  float v0 = xr[t], v1 = xr[t + 256], v2 = xr[t + 512];
  __shared__ float rs[256], rq[256];
  rs[t] = v0 + v1 + v2;
  rq[t] = v0 * v0 + v1 * v1 + v2 * v2;
  __syncthreads();
#pragma unroll
  for (int off = 128; off > 0; off >>= 1) {
    if (t < off) { rs[t] += rs[t + off]; rq[t] += rq[t + off]; }
    __syncthreads();
  }
  const float mu  = rs[0] * (1.0f / DIM);
  const float var = rq[0] * (1.0f / DIM) - mu * mu;
  const float inv = rsqrtf(var + 1e-5f);
  half_t* orow = out + (size_t)row * DIM;
  orow[t]       = (half_t)((v0 - mu) * inv * g[t]       + b[t]);
  orow[t + 256] = (half_t)((v1 - mu) * inv * g[t + 256] + b[t + 256]);
  orow[t + 512] = (half_t)((v2 - mu) * inv * g[t + 512] + b[t + 512]);
}

// ---------------- WMMA GEMM: C = A(MxK) * BT(NxK)^T + bias ------------------
// Block: 256 thr / 8 waves, tile 128x64, K-step 32, LDS double buffer.
// A tile (8 KB) via TDM (wave 0 issues, TENSORcnt); B tile via async-to-LDS.
// MODE 0: store f16; MODE 1: +resid, store f32; MODE 2: exact GELU, store f16.
template <int MODE>
__global__ __launch_bounds__(256)
void gemm_wmma_kernel(const half_t* __restrict__ A, const half_t* __restrict__ BT,
                      const float* __restrict__ bias, const float* __restrict__ resid,
                      float* __restrict__ outF, half_t* __restrict__ outH,
                      int M, int N, int K) {
  __shared__ half_t sA[2][128 * 32];
  __shared__ half_t sB[2][64 * 32];

  const int tid  = threadIdx.x;
  const int wave = tid >> 5;
  const int lane = tid & 31;
  const int rowA0 = blockIdx.y * 128;
  const int row0  = rowA0 + wave * 16;
  const int col0  = blockIdx.x * 64;

  const unsigned ldsA = (unsigned)(size_t)(&sA[0][0]);
  const unsigned ldsB = (unsigned)(size_t)(&sB[0][0]);

  // B staging coordinates: tile = 2048 halves -> 8 per thread (one b128).
  const int lb = tid * 8, rb = lb >> 5, cb = lb & 31;

  auto issueB = [&](int buf, int k) {
    const half_t* gb = BT + (size_t)(col0 + rb) * K + k + cb;
    unsigned ob = ldsB + (unsigned)(buf * (64 * 32) + lb) * 2u;
    async_b128(ob, gb);
  };
  auto issueA = [&](int buf, int k) {
    tdm_load_2d(ldsA + (unsigned)(buf * (128 * 32)) * 2u,
                A + (size_t)rowA0 * K + k,
                (unsigned)K, (unsigned)M, 32u, 128u, (unsigned)K);
  };

  const int steps = K >> 5;
  issueB(0, 0);
  if (wave == 0) issueA(0, 0);
  wait_async0();
  if (wave == 0) __builtin_amdgcn_s_wait_tensorcnt(0);
  __syncthreads();

  v8f acc[4] = {};
  const int m  = lane & 15;
  const int kb = (lane < 16) ? 0 : 8;    // A-frag K sub-offset
  const int ko = (lane < 16) ? 0 : 8;    // B-frag K sub-offset

  for (int ks = 0; ks < steps; ++ks) {
    const int cur = ks & 1;
    if (ks + 1 < steps) {
      issueB(cur ^ 1, (ks + 1) * 32);
      if (wave == 0) issueA(cur ^ 1, (ks + 1) * 32);
    }

    v16h a;
    {
      const int base = cur * (128 * 32) + (wave * 16 + m) * 32 + kb;
      v8h lo = *(const v8h*)(&sA[0][0] + base);
      v8h hi = *(const v8h*)(&sA[0][0] + base + 16);
#pragma unroll
      for (int i = 0; i < 8; ++i) { a[i] = lo[i]; a[8 + i] = hi[i]; }
    }
#pragma unroll
    for (int t = 0; t < 4; ++t) {
      const int base = cur * (64 * 32) + (16 * t + m) * 32 + ko;
      v8h lo = *(const v8h*)(&sB[0][0] + base);
      v8h hi = *(const v8h*)(&sB[0][0] + base + 16);
      v16h b;
#pragma unroll
      for (int i = 0; i < 8; ++i) { b[i] = lo[i]; b[8 + i] = hi[i]; }
      acc[t] = __builtin_amdgcn_wmma_f32_16x16x32_f16(
          false, a, false, b, (short)0, acc[t], false, false);
    }
    wait_async0();
    if (wave == 0) __builtin_amdgcn_s_wait_tensorcnt(0);
    __syncthreads();
  }

  const int mg = (lane < 16) ? 0 : 8;
#pragma unroll
  for (int t = 0; t < 4; ++t) {
    const int col = col0 + 16 * t + m;
    const float bv = bias[col];
#pragma unroll
    for (int r = 0; r < 8; ++r) {
      const size_t idx = (size_t)(row0 + mg + r) * N + col;
      float z = acc[t][r] + bv;
      if (MODE == 0) {
        outH[idx] = (half_t)z;
      } else if (MODE == 1) {
        outF[idx] = z + resid[idx];
      } else {
        float ge = 0.5f * z * (1.0f + erff(z * 0.70710678118654752f));
        outH[idx] = (half_t)ge;
      }
    }
  }
}

// ---------------- flash attention -------------------------------------------
// Block: 128 thr / 4 waves sharing one (b,h); wave handles one 16-row Q tile.
__global__ __launch_bounds__(128)
void flash_attn_kernel(const half_t* __restrict__ qkv, const float* __restrict__ mbias,
                       half_t* __restrict__ o16) {
  __shared__ half_t sVt[64 * 32];          // Vt[d][key], 4 KB

  const int tid  = threadIdx.x;
  const int wave = tid >> 5;
  const int lane = tid & 31;
  const int bh = blockIdx.x >> 4;          // 0..47
  const int qt = ((blockIdx.x & 15) << 2) + wave;
  const int b  = bh / HEADS;
  const int h  = bh - b * HEADS;

  const size_t RS = QKVC;
  const half_t* qbase = qkv + (size_t)(b * SEQ) * RS + h * HD;
  const half_t* kbase = qbase + DIM;
  const half_t* vbase = qbase + 2 * DIM;
  const float* brow = mbias + b * SEQ;

  const int nq   = lane & 15;
  const int grp8 = (lane < 16) ? 0 : 8;
  const int ko16 = (lane < 16) ? 0 : 8;

  // Q^T B-fragments (loop invariant), d-chunks 0 and 32.
  v16h qb[2];
#pragma unroll
  for (int dc = 0; dc < 2; ++dc) {
    const half_t* p = qbase + (size_t)(qt * 16 + nq) * RS + dc * 32 + ko16;
    v8h lo = *(const v8h*)(p);
    v8h hi = *(const v8h*)(p + 16);
#pragma unroll
    for (int i = 0; i < 8; ++i) { qb[dc][i] = lo[i]; qb[dc][8 + i] = hi[i]; }
  }

  float m_run = -1e30f, l_run = 0.0f;
  v8f oacc[4] = {};

  for (int kb2 = 0; kb2 < SEQ; kb2 += 32) {
    __syncthreads();                       // protect sVt from previous readers
    {
      const int keyl = tid >> 2;           // 0..31
      const int ch   = tid & 3;            // 0..3
      const half_t* vp = vbase + (size_t)(kb2 + keyl) * RS + ch * 16;
      v8h a0 = *(const v8h*)(vp);
      v8h a1 = *(const v8h*)(vp + 8);
#pragma unroll
      for (int i = 0; i < 8; ++i) {
        sVt[(ch * 16 + i) * 32 + keyl]     = a0[i];
        sVt[(ch * 16 + 8 + i) * 32 + keyl] = a1[i];
      }
    }
    __syncthreads();

    // S^T(32 keys x 16 q): A = K rows (contiguous global), B = qb.
    v8f st[2] = {};
#pragma unroll
    for (int t = 0; t < 2; ++t) {
      const half_t* krow = kbase + (size_t)(kb2 + 16 * t + nq) * RS + grp8;
#pragma unroll
      for (int dc = 0; dc < 2; ++dc) {
        const half_t* p = krow + dc * 32;
        v8h lo = *(const v8h*)(p);
        v8h hi = *(const v8h*)(p + 16);
        v16h a;
#pragma unroll
        for (int i = 0; i < 8; ++i) { a[i] = lo[i]; a[8 + i] = hi[i]; }
        st[t] = __builtin_amdgcn_wmma_f32_16x16x32_f16(
            false, a, false, qb[dc], (short)0, st[t], false, false);
      }
    }

    // Additive mask bias: 4 float4 loads (contiguous keys per lane group).
    union FU { float4 v4[2]; float f[8]; };
    FU fu[2];
    fu[0].v4[0] = *(const float4*)(brow + kb2 + grp8);
    fu[0].v4[1] = *(const float4*)(brow + kb2 + grp8 + 4);
    fu[1].v4[0] = *(const float4*)(brow + kb2 + 16 + grp8);
    fu[1].v4[1] = *(const float4*)(brow + kb2 + 16 + grp8 + 4);

    // Online softmax (row = q = lane column of S^T).
    float pv[2][8];
    float tmax = -1e30f;
#pragma unroll
    for (int t = 0; t < 2; ++t)
#pragma unroll
      for (int r = 0; r < 8; ++r) {
        float sv = st[t][r] * 0.125f + fu[t].f[r];
        pv[t][r] = sv;
        tmax = fmaxf(tmax, sv);
      }
    tmax = fmaxf(tmax, __shfl_xor(tmax, 16));
    const float m_new = fmaxf(m_run, tmax);
    const float alpha = __expf(m_run - m_new);
    float lsum = 0.0f;
#pragma unroll
    for (int t = 0; t < 2; ++t)
#pragma unroll
      for (int r = 0; r < 8; ++r) {
        float e = __expf(pv[t][r] - m_new);
        pv[t][r] = e;
        lsum += e;
      }
    lsum += __shfl_xor(lsum, 16);
    l_run = l_run * alpha + lsum;
    m_run = m_new;

    // Broadcast alpha into the O row domain (rows q = grp8+r).
    float alpha_r[8];
#pragma unroll
    for (int r = 0; r < 8; ++r) alpha_r[r] = __shfl(alpha, grp8 + r);
#pragma unroll
    for (int dt = 0; dt < 4; ++dt)
#pragma unroll
      for (int r = 0; r < 8; ++r) oacc[dt][r] *= alpha_r[r];

    // P fragment is already in WMMA A-layout: lane m=q, halves = own keys.
    v16h pa;
#pragma unroll
    for (int j = 0; j < 8; ++j) {
      pa[j]     = (half_t)pv[0][j];
      pa[8 + j] = (half_t)pv[1][j];
    }

    // O(16xq, 64d) += P(16x32) * V(32x64): B fragments from sVt, contiguous.
#pragma unroll
    for (int dt = 0; dt < 4; ++dt) {
      const int base = (dt * 16 + nq) * 32 + ko16;
      v8h lo = *(const v8h*)(&sVt[base]);
      v8h hi = *(const v8h*)(&sVt[base + 16]);
      v16h vb;
#pragma unroll
      for (int i = 0; i < 8; ++i) { vb[i] = lo[i]; vb[8 + i] = hi[i]; }
      oacc[dt] = __builtin_amdgcn_wmma_f32_16x16x32_f16(
          false, pa, false, vb, (short)0, oacc[dt], false, false);
    }
  }

  // Normalize (1/l per O row) and store.
  float linv[8];
#pragma unroll
  for (int r = 0; r < 8; ++r) linv[r] = 1.0f / __shfl(l_run, grp8 + r);
#pragma unroll
  for (int dt = 0; dt < 4; ++dt)
#pragma unroll
    for (int r = 0; r < 8; ++r)
      o16[(size_t)(b * SEQ + qt * 16 + grp8 + r) * DIM + h * HD + dt * 16 + nq] =
          (half_t)(oacc[dt][r] * linv[r]);
}

// ---------------------------------------------------------------------------
extern "C" void kernel_launch(void* const* d_in, const int* in_sizes, int n_in,
                              void* d_out, int out_size, void* d_ws, size_t ws_size,
                              hipStream_t stream) {
  const float* x      = (const float*)d_in[0];
  const int*   amask  = (const int*)d_in[1];
  const float* ln1_g  = (const float*)d_in[2];
  const float* ln1_b  = (const float*)d_in[3];
  const float* ln2_g  = (const float*)d_in[4];
  const float* ln2_b  = (const float*)d_in[5];
  const float* w_qkv  = (const float*)d_in[6];
  const float* b_qkv  = (const float*)d_in[7];
  const float* w_proj = (const float*)d_in[8];
  const float* b_proj = (const float*)d_in[9];
  const float* w_fc1  = (const float*)d_in[10];
  const float* b_fc1  = (const float*)d_in[11];
  const float* w_fc2  = (const float*)d_in[12];
  const float* b_fc2  = (const float*)d_in[13];

  char* ws = (char*)d_ws;
  size_t off = 0;
  auto alloc = [&](size_t bytes) -> char* {
    off = (off + 255) & ~(size_t)255;
    char* p = ws + off;
    off += bytes;
    return p;
  };
  half_t* h16    = (half_t*)alloc((size_t)ROWS * DIM * 2);
  half_t* wqkvT  = (half_t*)alloc((size_t)QKVC * DIM * 2);
  half_t* wprojT = (half_t*)alloc((size_t)DIM * DIM * 2);
  half_t* wfc1T  = (half_t*)alloc((size_t)HIDDEN * DIM * 2);
  half_t* wfc2T  = (half_t*)alloc((size_t)DIM * HIDDEN * 2);
  half_t* qkv16  = (half_t*)alloc((size_t)ROWS * QKVC * 2);
  half_t* o16    = (half_t*)alloc((size_t)ROWS * DIM * 2);
  float*  x1     = (float*)alloc((size_t)ROWS * DIM * 4);
  half_t* h2     = (half_t*)alloc((size_t)ROWS * DIM * 2);
  half_t* g16    = (half_t*)alloc((size_t)ROWS * HIDDEN * 2);
  float*  mbias  = (float*)alloc((size_t)ROWS * 4);

  cast_transpose_kernel<<<((size_t)DIM * QKVC + 255) / 256, 256, 0, stream>>>(w_qkv, wqkvT, DIM, QKVC);
  cast_transpose_kernel<<<((size_t)DIM * DIM + 255) / 256, 256, 0, stream>>>(w_proj, wprojT, DIM, DIM);
  cast_transpose_kernel<<<((size_t)DIM * HIDDEN + 255) / 256, 256, 0, stream>>>(w_fc1, wfc1T, DIM, HIDDEN);
  cast_transpose_kernel<<<((size_t)HIDDEN * DIM + 255) / 256, 256, 0, stream>>>(w_fc2, wfc2T, HIDDEN, DIM);
  mask_bias_kernel<<<ROWS / 256, 256, 0, stream>>>(amask, mbias);

  layernorm_f16_kernel<<<ROWS, 256, 0, stream>>>(x, ln1_g, ln1_b, h16);
  gemm_wmma_kernel<0><<<dim3(QKVC / 64, ROWS / 128), 256, 0, stream>>>(
      h16, wqkvT, b_qkv, nullptr, nullptr, qkv16, ROWS, QKVC, DIM);
  flash_attn_kernel<<<BATCH * HEADS * (SEQ / 64), 128, 0, stream>>>(qkv16, mbias, o16);
  gemm_wmma_kernel<1><<<dim3(DIM / 64, ROWS / 128), 256, 0, stream>>>(
      o16, wprojT, b_proj, x, x1, nullptr, ROWS, DIM, DIM);
  layernorm_f16_kernel<<<ROWS, 256, 0, stream>>>(x1, ln2_g, ln2_b, h2);
  gemm_wmma_kernel<2><<<dim3(HIDDEN / 64, ROWS / 128), 256, 0, stream>>>(
      h2, wfc1T, b_fc1, nullptr, nullptr, g16, ROWS, HIDDEN, DIM);
  gemm_wmma_kernel<1><<<dim3(DIM / 64, ROWS / 128), 256, 0, stream>>>(
      g16, wfc2T, b_fc2, x1, (float*)d_out, nullptr, ROWS, DIM, HIDDEN);
}
